// WaveModel_41162966564996
// MI455X (gfx1250) — matrix operand order repaired and checked by reading, not compiled, not run on hardware
//
#include <hip/hip_runtime.h>

typedef __attribute__((ext_vector_type(16))) _Float16 v16h;
typedef __attribute__((ext_vector_type(8)))  float     v8f;

namespace {

constexpr int HDIM = 64;          // LSTM hidden size
constexpr int G4   = 256;         // 4*H gate width
constexpr int BT   = 16;          // batch rows per workgroup (= WMMA M)
constexpr int NTHREADS = 512;     // 16 waves: one N-tile per wave in GEMM phase
constexpr int GSTRIDE = G4 + 4;   // padded gate-row stride (floats)
constexpr int HSTRIDE = HDIM + 8; // padded h-row stride (halves)

constexpr float LOG2E = 1.44269504088896340736f;

__device__ __forceinline__ float fast_rcp(float x)  { return __builtin_amdgcn_rcpf(x); }
__device__ __forceinline__ float fast_exp2(float x) { return __builtin_amdgcn_exp2f(x); }

// Branchless logistic family: scale * sigmoid-like + bias.
//   sigmoid(x) = rcp(1 + 2^(-log2e * x))                (k=-log2e, s=1, b=0)
//   tanh(x)    = 2*rcp(1 + 2^(-2*log2e * x)) - 1        (k=-2log2e, s=2, b=-1)
__device__ __forceinline__ float logistic(float x, float k, float s, float b) {
    float e = fast_exp2(x * k);
    return fmaf(s, fast_rcp(1.0f + e), b);
}
__device__ __forceinline__ float tanh_fast(float x) {
    return logistic(x, -2.0f * LOG2E, 2.0f, -1.0f);
}

// B-fragment for V_WMMA_F32_16X16X32_F16.  B = W^T (K x 16).  Per ISA B layout:
// lane holds column n = lane%16; lanes 0-15 hold K={koff..koff+7, koff+16..koff+23},
// lanes 16-31 shifted by +8 (folded into koff).  Column n of W^T is row n of W.
__device__ __forceinline__ v16h load_bfrag(const float* __restrict__ W, int n, int koff) {
    const float* p = W + n * HDIM + koff;
    v16h b;
#pragma unroll
    for (int j = 0; j < 8; ++j) { b[j] = (_Float16)p[j]; b[j + 8] = (_Float16)p[j + 16]; }
    return b;
}

// A-fragment (16 x 32 f16) from LDS h-state.  Per ISA A layout: lane holds row
// m = lane%16; lanes 0-15: K = {koff..koff+7, koff+16..koff+23}; lanes 16-31
// are shifted by +8 (folded into koff by caller).  Two ds_load_b128.
__device__ __forceinline__ v16h load_afrag(const _Float16* hbase, int m, int koff) {
    const _Float16* p = hbase + m * HSTRIDE + koff;
    v16h a;
#pragma unroll
    for (int j = 0; j < 8; ++j) { a[j] = p[j]; a[j + 8] = p[j + 16]; }
    return a;
}

} // namespace

__global__ __launch_bounds__(NTHREADS)
void lstm2_fused_kernel(const float* __restrict__ x,      // [B,T]
                        const float* __restrict__ h0in,   // [2,B,H]
                        const float* __restrict__ c0in,   // [2,B,H]
                        const float* __restrict__ Wih0,   // [4H,1]
                        const float* __restrict__ Whh0,   // [4H,H]
                        const float* __restrict__ bih0,   // [4H]
                        const float* __restrict__ bhh0,   // [4H]
                        const float* __restrict__ Wih1,   // [4H,H]
                        const float* __restrict__ Whh1,   // [4H,H]
                        const float* __restrict__ bih1,   // [4H]
                        const float* __restrict__ bhh1,   // [4H]
                        const float* __restrict__ Wlin,   // [1,H]
                        const float* __restrict__ blin,   // [1]
                        float* __restrict__ out,          // [B,T]
                        int T, int Btotal) {
    __shared__ float    s_gates[BT * GSTRIDE];  // activated gates, 16 x 256 (padded)
    __shared__ _Float16 s_h0[BT * HSTRIDE];     // layer-0 h state, f16
    __shared__ _Float16 s_h1[BT * HSTRIDE];     // layer-1 h state, f16
    __shared__ float    s_x[BT];                // current-step scalar inputs

    const int tid  = threadIdx.x;
    const int lane = tid & 31;
    const int wv   = tid >> 5;          // wave id 0..15
    const int l16  = lane & 15;
    const int hi   = lane >> 4;         // lane half
    const int row0 = blockIdx.x * BT;   // first global batch row of this WG

    // ---- per-lane GEMM-phase constants (wave wv owns gate columns [16wv,16wv+16))
    const int n = wv * 16 + l16;        // owned gate column
    const float bsum0 = bih0[n] + bhh0[n];
    const float bsum1 = bih1[n] + bhh1[n];
    const float wih0c = Wih0[n];        // layer-0 input projection (D==1)

    const int kb0 = hi * 8;             // B-frag lane K offset, k-step 0
    const int kb1 = 32 + hi * 8;        // k-step 1
    const v16h bh0_0 = load_bfrag(Whh0, n, kb0);  // weights pinned in VGPRs,
    const v16h bh0_1 = load_bfrag(Whh0, n, kb1);  // converted to f16 once and
    const v16h bh1_0 = load_bfrag(Whh1, n, kb0);  // reused for all 2048 steps
    const v16h bh1_1 = load_bfrag(Whh1, n, kb1);
    const v16h bx1_0 = load_bfrag(Wih1, n, kb0);
    const v16h bx1_1 = load_bfrag(Wih1, n, kb1);

    // ---- per-thread update-phase constants (wave <-> batch row, lane <-> 2 cols)
    const int um = wv;
    const int uj = lane * 2;
    const int gb = row0 + um;           // global batch row
    float c0a = c0in[0 * Btotal * HDIM + gb * HDIM + uj];
    float c0b = c0in[0 * Btotal * HDIM + gb * HDIM + uj + 1];
    float c1a = c0in[1 * Btotal * HDIM + gb * HDIM + uj];
    float c1b = c0in[1 * Btotal * HDIM + gb * HDIM + uj + 1];
    const float wl0 = Wlin[uj], wl1 = Wlin[uj + 1];
    const float bl  = blin[0];

    s_h0[um * HSTRIDE + uj]     = (_Float16)h0in[0 * Btotal * HDIM + gb * HDIM + uj];
    s_h0[um * HSTRIDE + uj + 1] = (_Float16)h0in[0 * Btotal * HDIM + gb * HDIM + uj + 1];
    s_h1[um * HSTRIDE + uj]     = (_Float16)h0in[1 * Btotal * HDIM + gb * HDIM + uj];
    s_h1[um * HSTRIDE + uj + 1] = (_Float16)h0in[1 * Btotal * HDIM + gb * HDIM + uj + 1];
    if (wv == 0 && lane < BT) s_x[lane] = x[(row0 + lane) * T + 0];

    const int akoff = hi * 8;           // A-frag lane K offset
    const int mb    = hi * 8;           // first C/D row for this lane half
    const bool isG  = (wv >> 2) == 2;   // columns [128,192) hold the 'g' gate

    // Branchless per-wave activation constants: 'g' gate uses tanh, others sigmoid.
    const float gk = isG ? (-2.0f * LOG2E) : (-LOG2E);  // exp2 coefficient
    const float gs = isG ? 2.0f : 1.0f;                 // output scale
    const float go = isG ? -1.0f : 0.0f;                // output bias

    __syncthreads();

    for (int t = 0; t < T; ++t) {
        // ---- Phase A: layer-0 gates = h0 @ Whh0^T + x*Wih0 + (bih0+bhh0) ----
        {
            v8f acc;
#pragma unroll
            for (int v = 0; v < 8; ++v) acc[v] = bsum0 + s_x[mb + v] * wih0c;
            v16h a0 = load_afrag(s_h0, l16, akoff);
            v16h a1 = load_afrag(s_h0, l16, 32 + akoff);
            acc = __builtin_amdgcn_wmma_f32_16x16x32_f16(false, a0, false, bh0_0,
                                                         (short)0, acc, false, false);
            acc = __builtin_amdgcn_wmma_f32_16x16x32_f16(false, a1, false, bh0_1,
                                                         (short)0, acc, false, false);
#pragma unroll
            for (int v = 0; v < 8; ++v)
                s_gates[(mb + v) * GSTRIDE + n] = logistic(acc[v], gk, gs, go);
        }
        __syncthreads();

        // ---- Phase B: layer-0 cell update (c in registers, h -> LDS f16) ----
        {
            const float* gr = &s_gates[um * GSTRIDE];
            float i0 = gr[uj],       i1 = gr[uj + 1];
            float f0 = gr[64 + uj],  f1 = gr[64 + uj + 1];
            float g0 = gr[128 + uj], g1 = gr[128 + uj + 1];
            float o0 = gr[192 + uj], o1 = gr[192 + uj + 1];
            c0a = fmaf(f0, c0a, i0 * g0);
            c0b = fmaf(f1, c0b, i1 * g1);
            s_h0[um * HSTRIDE + uj]     = (_Float16)(o0 * tanh_fast(c0a));
            s_h0[um * HSTRIDE + uj + 1] = (_Float16)(o1 * tanh_fast(c0b));
        }
        __syncthreads();

        // ---- Phase C: layer-1 gates = h1 @ Whh1^T + h0 @ Wih1^T + biases ----
        {
            v8f acc;
#pragma unroll
            for (int v = 0; v < 8; ++v) acc[v] = bsum1;
            v16h a0 = load_afrag(s_h1, l16, akoff);
            v16h a1 = load_afrag(s_h1, l16, 32 + akoff);
            acc = __builtin_amdgcn_wmma_f32_16x16x32_f16(false, a0, false, bh1_0,
                                                         (short)0, acc, false, false);
            acc = __builtin_amdgcn_wmma_f32_16x16x32_f16(false, a1, false, bh1_1,
                                                         (short)0, acc, false, false);
            v16h a2 = load_afrag(s_h0, l16, akoff);
            v16h a3 = load_afrag(s_h0, l16, 32 + akoff);
            acc = __builtin_amdgcn_wmma_f32_16x16x32_f16(false, a2, false, bx1_0,
                                                         (short)0, acc, false, false);
            acc = __builtin_amdgcn_wmma_f32_16x16x32_f16(false, a3, false, bx1_1,
                                                         (short)0, acc, false, false);
#pragma unroll
            for (int v = 0; v < 8; ++v)
                s_gates[(mb + v) * GSTRIDE + n] = logistic(acc[v], gk, gs, go);
        }
        __syncthreads();

        // ---- Phase D: layer-1 update + linear head + x(t+1) prefetch --------
        {
            const float* gr = &s_gates[um * GSTRIDE];
            float i0 = gr[uj],       i1 = gr[uj + 1];
            float f0 = gr[64 + uj],  f1 = gr[64 + uj + 1];
            float g0 = gr[128 + uj], g1 = gr[128 + uj + 1];
            float o0 = gr[192 + uj], o1 = gr[192 + uj + 1];
            c1a = fmaf(f0, c1a, i0 * g0);
            c1b = fmaf(f1, c1b, i1 * g1);
            float h1a = o0 * tanh_fast(c1a);
            float h1b = o1 * tanh_fast(c1b);
            s_h1[um * HSTRIDE + uj]     = (_Float16)h1a;
            s_h1[um * HSTRIDE + uj + 1] = (_Float16)h1b;

            // output head: dot(h1[row], Wlin) reduced across the wave (wave32)
            float p = fmaf(h1a, wl0, h1b * wl1);
#pragma unroll
            for (int off = 16; off > 0; off >>= 1) p += __shfl_xor(p, off, 32);
            if (lane == 0) out[gb * T + t] = p + bl;

            if (wv == 0 && lane < BT && t + 1 < T)
                s_x[lane] = x[(row0 + lane) * T + (t + 1)];
        }
        __syncthreads();
    }
}

extern "C" void kernel_launch(void* const* d_in, const int* in_sizes, int n_in,
                              void* d_out, int out_size, void* d_ws, size_t ws_size,
                              hipStream_t stream) {
    (void)n_in; (void)d_ws; (void)ws_size; (void)out_size;
    const float* x    = (const float*)d_in[0];
    const float* h0   = (const float*)d_in[1];
    const float* c0   = (const float*)d_in[2];
    const float* Wih0 = (const float*)d_in[3];
    const float* Whh0 = (const float*)d_in[4];
    const float* bih0 = (const float*)d_in[5];
    const float* bhh0 = (const float*)d_in[6];
    const float* Wih1 = (const float*)d_in[7];
    const float* Whh1 = (const float*)d_in[8];
    const float* bih1 = (const float*)d_in[9];
    const float* bhh1 = (const float*)d_in[10];
    const float* Wlin = (const float*)d_in[11];
    const float* blin = (const float*)d_in[12];
    float* out = (float*)d_out;

    const int Btotal = 256;
    const int T = in_sizes[0] / Btotal;   // 2048

    dim3 grid(Btotal / BT);               // 16 persistent workgroups
    dim3 block(NTHREADS);                 // 16 waves (wave32)
    hipLaunchKernelGGL(lstm2_fused_kernel, grid, block, 0, stream,
                       x, h0, c0, Wih0, Whh0, bih0, bhh0,
                       Wih1, Whh1, bih1, bhh1, Wlin, blin, out, T, Btotal);
}
